// PoseGAT_86002425135776
// MI455X (gfx1250) — matrix operand order repaired and checked by reading, not compiled, not run on hardware
//
#include <hip/hip_runtime.h>
#include <hip/hip_bf16.h>
#include <stdint.h>

// ---------------------------------------------------------------------------
// Types for CDNA5 WMMA (wave32)
// ---------------------------------------------------------------------------
typedef __attribute__((ext_vector_type(16))) __bf16       v16bf;
typedef __attribute__((ext_vector_type(8)))  float        v8f;
typedef __attribute__((ext_vector_type(4)))  unsigned int v4u;

union Frag { v16bf b; v4u u[2]; };

__device__ __forceinline__ unsigned short f2bf(float f) {
    // round-to-nearest-even fp32 -> bf16
    unsigned u = __float_as_uint(f);
    unsigned r = u + 0x7FFFu + ((u >> 16) & 1u);
    return (unsigned short)(r >> 16);
}

__device__ __forceinline__ unsigned fkey(float f) {
    // order-preserving float -> u32 key (monotone); key(x) > 0 for all reals
    unsigned u = __float_as_uint(f);
    return (u & 0x80000000u) ? ~u : (u | 0x80000000u);
}
__device__ __forceinline__ float funkey(unsigned k) {
    unsigned u = (k & 0x80000000u) ? (k & 0x7FFFFFFFu) : ~k;
    return __uint_as_float(u);
}

// ---------------------------------------------------------------------------
// Zero a u32-granular region
// ---------------------------------------------------------------------------
__global__ void k_zero(unsigned* __restrict__ p, size_t n) {
    size_t t = (size_t)blockIdx.x * blockDim.x + threadIdx.x;
    if (t < n) p[t] = 0u;
}

// ---------------------------------------------------------------------------
// fp32 -> bf16 elementwise conversion (activations)
// ---------------------------------------------------------------------------
__global__ void k_convbf(const float* __restrict__ src, unsigned short* __restrict__ dst, size_t n) {
    size_t t = (size_t)blockIdx.x * blockDim.x + threadIdx.x;
    if (t < n) dst[t] = f2bf(src[t]);
}

// ---------------------------------------------------------------------------
// Pack W (f32, [K,M] row-major) into fragment-ready bf16:
// Wp[(((ct*Ktiles)+kt)*32 + lane)*16 + j] = bf16(W[(kt*32 + half*8 + j + (j>>3)*8)*M + ct*16 + (lane&15)])
// so each lane's 16 B-fragment values are one contiguous 32-byte run.
// ---------------------------------------------------------------------------
__global__ void k_packw(const float* __restrict__ W, unsigned short* __restrict__ Wp,
                        int K, int M) {
    int t = blockIdx.x * blockDim.x + threadIdx.x;
    int Ktiles = K >> 5;
    int total  = Ktiles * (M >> 4) * 512;
    if (t >= total) return;
    int j    = t & 15;
    int l    = (t >> 4) & 31;
    int tile = t >> 9;               // ct*Ktiles + kt
    int ct   = tile / Ktiles;
    int kt   = tile % Ktiles;
    int c = l & 15, half = l >> 4;
    int Kidx = kt * 32 + half * 8 + j + (j >> 3) * 8;
    int col  = ct * 16 + c;
    Wp[t] = f2bf(W[(size_t)Kidx * M + col]);
}

// ---------------------------------------------------------------------------
// WMMA GEMM: Y[N,M] = Xb[N,K](bf16) @ Wp(packed bf16), f32 accumulate.
// One 16x16 output tile per wave32; K-loop fully unrolled.
// A layout (ISA 7.12.2, 16-bit A 16x32): lane half 0 -> K in {0..7,16..23},
// half 1 -> K in {8..15,24..31}, row = lane&15.  Two b128 loads per step.
// B comes pre-packed: two b128 loads per step, L2-resident across row tiles.
// C/D: VGPR g -> row g + half*8, col = lane&15.
// Full tiles (wave-uniform test) take a divergence-free fast path.
// ---------------------------------------------------------------------------
template<int K, int M>
__global__ void k_gemm_wmma(const unsigned short* __restrict__ Xb,
                            const unsigned short* __restrict__ Wp,
                            float* __restrict__ Y, int N) {
    constexpr int Ktiles   = K >> 5;
    constexpr int colTiles = M >> 4;
    int wave = (blockIdx.x * blockDim.x + threadIdx.x) >> 5;
    int lane = threadIdx.x & 31;
    int rowTiles = (N + 15) >> 4;
    if (wave >= rowTiles * colTiles) return;   // wave-uniform: EXEC all-1 in live waves
    int rt = wave % rowTiles;
    int ct = wave / rowTiles;

    int r = lane & 15, half = lane >> 4;
    int row = rt * 16 + r;
    int col = ct * 16 + r;

    const unsigned short* xrow  = Xb + (size_t)row * K;
    const unsigned short* wbase = Wp + ((size_t)ct * Ktiles * 32 + lane) * 16;

    v8f acc = {};
    if (rt * 16 + 16 <= N) {
        // ---- fast path: full 16-row tile, wave-uniform, no exec masking ----
#pragma unroll
        for (int kt = 0; kt < Ktiles; ++kt) {
            Frag a, b;
            a.u[0] = *(const v4u*)(xrow + kt * 32 + half * 8);
            a.u[1] = *(const v4u*)(xrow + kt * 32 + 16 + half * 8);
            const unsigned short* wq = wbase + (size_t)kt * 512;
            b.u[0] = *(const v4u*)(wq);
            b.u[1] = *(const v4u*)(wq + 8);
            acc = __builtin_amdgcn_wmma_f32_16x16x32_bf16(
                      false, a.b, false, b.b, (short)0, acc, false, false);
        }
        float* yb = Y + (size_t)(rt * 16 + half * 8) * M + col;
#pragma unroll
        for (int g = 0; g < 8; ++g) yb[(size_t)g * M] = acc[g];
    } else {
        // ---- tail tile: per-lane guards ----
        bool inr = row < N;
#pragma unroll
        for (int kt = 0; kt < Ktiles; ++kt) {
            Frag a, b;
            if (inr) {
                a.u[0] = *(const v4u*)(xrow + kt * 32 + half * 8);
                a.u[1] = *(const v4u*)(xrow + kt * 32 + 16 + half * 8);
            } else {
                a.u[0] = (v4u){0u, 0u, 0u, 0u};
                a.u[1] = (v4u){0u, 0u, 0u, 0u};
            }
            const unsigned short* wq = wbase + (size_t)kt * 512;
            b.u[0] = *(const v4u*)(wq);
            b.u[1] = *(const v4u*)(wq + 8);
            acc = __builtin_amdgcn_wmma_f32_16x16x32_bf16(
                      false, a.b, false, b.b, (short)0, acc, false, false);
        }
#pragma unroll
        for (int g = 0; g < 8; ++g) {
            int orow = rt * 16 + g + half * 8;
            if (orow < N) Y[(size_t)orow * M + col] = acc[g];
        }
    }
}

// ---------------------------------------------------------------------------
// Per-node attention scores: a_s[n,h] = <feat[n,h,:], att_src[h,:]> (C = 32)
// ---------------------------------------------------------------------------
__global__ void k_att(const float* __restrict__ feat,
                      const float* __restrict__ att_s, const float* __restrict__ att_d,
                      float* __restrict__ as_o, float* __restrict__ ad_o,
                      int N, int H, int C) {
    int t = blockIdx.x * blockDim.x + threadIdx.x;
    if (t >= N * H) return;
    int n = t / H, h = t % H;
    const float* f = feat + (size_t)n * H * C + h * C;
    const float* s = att_s + h * C;
    const float* d = att_d + h * C;
    float vs = 0.f, vd = 0.f;
    for (int c = 0; c < C; ++c) { float x = f[c]; vs += x * s[c]; vd += x * d[c]; }
    as_o[t] = vs;
    ad_o[t] = vd;
}

__device__ __forceinline__ void edge_sd(const long long* __restrict__ ei, long long E,
                                        long long e, int& s, int& d) {
    if (e < E) { s = (int)ei[e]; d = (int)ei[E + e]; }
    else       { s = d = (int)(e - E); }           // appended self-loops
}

// ---------------------------------------------------------------------------
// Segment-softmax pass 1: running max per (dst, head) via ordered-key atomicMax
// ---------------------------------------------------------------------------
__global__ void k_edge_max(const long long* __restrict__ ei, long long E, int n, int H,
                           const float* __restrict__ as_, const float* __restrict__ ad_,
                           unsigned* __restrict__ maxb) {
    long long t = (long long)blockIdx.x * blockDim.x + threadIdx.x;
    long long Etot = E + n;
    if (t >= Etot * H) return;
    long long e = t / H; int h = (int)(t % H);
    int s, d; edge_sd(ei, E, e, s, d);
    float el = as_[(size_t)s * H + h] + ad_[(size_t)d * H + h];
    el = el > 0.f ? el : 0.2f * el;                // leaky_relu(0.2)
    atomicMax(&maxb[(size_t)d * H + h], fkey(el));
}

// ---------------------------------------------------------------------------
// Segment-softmax pass 2: denom[dst,h] += exp(e - max)
// ---------------------------------------------------------------------------
__global__ void k_edge_sum(const long long* __restrict__ ei, long long E, int n, int H,
                           const float* __restrict__ as_, const float* __restrict__ ad_,
                           const unsigned* __restrict__ maxb, float* __restrict__ denom) {
    long long t = (long long)blockIdx.x * blockDim.x + threadIdx.x;
    long long Etot = E + n;
    if (t >= Etot * H) return;
    long long e = t / H; int h = (int)(t % H);
    int s, d; edge_sd(ei, E, e, s, d);
    float el = as_[(size_t)s * H + h] + ad_[(size_t)d * H + h];
    el = el > 0.f ? el : 0.2f * el;
    float m = funkey(maxb[(size_t)d * H + h]);
    atomicAdd(&denom[(size_t)d * H + h], __expf(el - m));
}

// ---------------------------------------------------------------------------
// Pass 3: agg[dst,:] += alpha(e,h) * feat[src,:]; one wave32 per edge.
// alpha recomputed (as/ad/max/denom are L2-resident) to avoid a 26MB buffer.
// ---------------------------------------------------------------------------
__global__ void k_edge_agg(const long long* __restrict__ ei, long long E, int n, int H, int C,
                           const float* __restrict__ as_, const float* __restrict__ ad_,
                           const unsigned* __restrict__ maxb, const float* __restrict__ denom,
                           const float* __restrict__ feat, float* __restrict__ agg) {
    int lane = threadIdx.x & 31;
    long long e = ((long long)blockIdx.x * blockDim.x + threadIdx.x) >> 5;
    long long Etot = E + n;
    if (e >= Etot) return;
    int s, d; edge_sd(ei, E, e, s, d);
    int M = H * C;
    for (int i = lane; i < M; i += 32) {
        int h = i / C;                              // C==32 -> wave-uniform per iteration
        float el = as_[(size_t)s * H + h] + ad_[(size_t)d * H + h];
        el = el > 0.f ? el : 0.2f * el;
        float m = funkey(maxb[(size_t)d * H + h]);
        float alpha = __expf(el - m) / (denom[(size_t)d * H + h] + 1e-16f);
        atomicAdd(&agg[(size_t)d * M + i], feat[(size_t)s * M + i] * alpha);
    }
}

// ---------------------------------------------------------------------------
// out = agg + bias, optional ELU
// ---------------------------------------------------------------------------
__global__ void k_finalize(const float* __restrict__ agg, const float* __restrict__ bias,
                           float* __restrict__ out, size_t total, int M, int do_elu) {
    size_t t = (size_t)blockIdx.x * blockDim.x + threadIdx.x;
    if (t >= total) return;
    float v = agg[t] + bias[(int)(t % (size_t)M)];
    if (do_elu) v = v > 0.f ? v : (__expf(v) - 1.f);
    out[t] = v;
}

// ---------------------------------------------------------------------------
// Host-side orchestration
// ---------------------------------------------------------------------------
static inline unsigned gdiv(long long a, long long b) { return (unsigned)((a + b - 1) / b); }

extern "C" void kernel_launch(void* const* d_in, const int* in_sizes, int n_in,
                              void* d_out, int out_size, void* d_ws, size_t ws_size,
                              hipStream_t stream) {
    const float*     x   = (const float*)d_in[0];
    const long long* ei  = (const long long*)d_in[1];
    const float*     W1  = (const float*)d_in[2];
    const float*     as1 = (const float*)d_in[3];
    const float*     ad1 = (const float*)d_in[4];
    const float*     b1  = (const float*)d_in[5];
    const float*     W2  = (const float*)d_in[6];
    const float*     as2 = (const float*)d_in[7];
    const float*     ad2 = (const float*)d_in[8];
    const float*     b2  = (const float*)d_in[9];
    const float*     W3  = (const float*)d_in[10];
    const float*     as3 = (const float*)d_in[11];
    const float*     ad3 = (const float*)d_in[12];
    const float*     b3  = (const float*)d_in[13];

    const int       IN = 64, HID = 32, OUT = 32, H = 4;
    const int       N  = in_sizes[0] / IN;
    const long long E  = (long long)in_sizes[1] / 2;
    const long long Etot = E + N;

    // workspace carve-up (all 16B-aligned by construction)
    char* w = (char*)d_ws;
    float* bufA = (float*)w; w += (size_t)N * 128 * sizeof(float);   // gemm out / feat
    float* bufB = (float*)w; w += (size_t)N * 128 * sizeof(float);   // aggregation
    float* bufC = (float*)w; w += (size_t)N * 128 * sizeof(float);   // layer input (f32)
    float*    asb   = (float*)w;    w += (size_t)N * H * sizeof(float);
    float*    adb   = (float*)w;    w += (size_t)N * H * sizeof(float);
    unsigned* maxb  = (unsigned*)w; w += (size_t)N * H * sizeof(unsigned);
    float*    denom = (float*)w;    w += (size_t)N * H * sizeof(float);
    unsigned short* Xbf = (unsigned short*)w; w += (size_t)N * 128 * sizeof(unsigned short);
    unsigned short* Wp  = (unsigned short*)w;                         // <= 128*128 bf16

    const int TB = 256;

    // Edge/softmax/aggregate/finalize phase (identical across layers)
    auto edge_phase = [&](const float* att_s, const float* att_d, const float* bias,
                          int heads, int Ml, float* outp, int do_elu) {
        k_zero<<<gdiv((long long)N * Ml, TB), TB, 0, stream>>>((unsigned*)bufB, (size_t)N * Ml);
        k_zero<<<gdiv((long long)N * heads, TB), TB, 0, stream>>>(maxb, (size_t)N * heads);
        k_zero<<<gdiv((long long)N * heads, TB), TB, 0, stream>>>((unsigned*)denom, (size_t)N * heads);
        k_att<<<gdiv((long long)N * heads, TB), TB, 0, stream>>>(bufA, att_s, att_d, asb, adb, N, heads, 32);
        k_edge_max<<<gdiv(Etot * heads, TB), TB, 0, stream>>>(ei, E, N, heads, asb, adb, maxb);
        k_edge_sum<<<gdiv(Etot * heads, TB), TB, 0, stream>>>(ei, E, N, heads, asb, adb, maxb, denom);
        k_edge_agg<<<gdiv(Etot * 32, TB), TB, 0, stream>>>(ei, E, N, heads, 32, asb, adb, maxb, denom, bufA, bufB);
        k_finalize<<<gdiv((long long)N * Ml, TB), TB, 0, stream>>>(bufB, bias, outp, (size_t)N * Ml, Ml, do_elu);
    };

    const long long rowTiles = (N + 15) / 16;

    // -------- Layer 1: [N,64] -> [N,128], 4 heads, concat, ELU --------
    k_convbf<<<gdiv((long long)N * 64, TB), TB, 0, stream>>>(x, Xbf, (size_t)N * 64);
    k_packw<<<gdiv(64 * 128, TB), TB, 0, stream>>>(W1, Wp, 64, 128);
    k_gemm_wmma<64, 128><<<gdiv(rowTiles * 8 * 32, 128), 128, 0, stream>>>(Xbf, Wp, bufA, N);
    edge_phase(as1, ad1, b1, H, 128, bufC, 1);

    // -------- Layer 2: [N,128] -> [N,128], 4 heads, concat, ELU --------
    k_convbf<<<gdiv((long long)N * 128, TB), TB, 0, stream>>>(bufC, Xbf, (size_t)N * 128);
    k_packw<<<gdiv(128 * 128, TB), TB, 0, stream>>>(W2, Wp, 128, 128);
    k_gemm_wmma<128, 128><<<gdiv(rowTiles * 8 * 32, 128), 128, 0, stream>>>(Xbf, Wp, bufA, N);
    edge_phase(as2, ad2, b2, H, 128, bufC, 1);

    // -------- Layer 3: [N,128] -> [N,32], 1 head, mean(=id), bias only -> d_out --------
    k_convbf<<<gdiv((long long)N * 128, TB), TB, 0, stream>>>(bufC, Xbf, (size_t)N * 128);
    k_packw<<<gdiv(128 * 32, TB), TB, 0, stream>>>(W3, Wp, 128, 32);
    k_gemm_wmma<128, 32><<<gdiv(rowTiles * 2 * 32, 128), 128, 0, stream>>>(Xbf, Wp, bufA, N);
    edge_phase(as3, ad3, b3, 1, 32, (float*)d_out, 0);
}
